// SeedOssAttention_67010079752789
// MI455X (gfx1250) — matrix operand order repaired and checked by reading, not compiled, not run on hardware
//
#include <hip/hip_runtime.h>

typedef __attribute__((ext_vector_type(16))) _Float16 v16h;
typedef __attribute__((ext_vector_type(8)))  _Float16 v8h;
typedef __attribute__((ext_vector_type(4)))  _Float16 v4h;
typedef __attribute__((ext_vector_type(8)))  float    v8f;
typedef int i32x4 __attribute__((vector_size(16)));   // b128 payload type

#define AS1 __attribute__((address_space(1)))
#define AS3 __attribute__((address_space(3)))

#if __has_builtin(__builtin_amdgcn_global_load_async_to_lds_b128)
#define USE_ASYNC_LDS 1
#endif

namespace {
constexpr int kB   = 2;
constexpr int kS   = 1024;
constexpr int kD   = 5120;
constexpr int kNH  = 80;
constexpr int kNKV = 8;
constexpr int kHD  = 128;
constexpr int kGroups = kNH / kNKV;     // 10
constexpr int kM   = kB * kS;           // 2048
constexpr int kNQ  = kNH * kHD;         // 10240
constexpr int kNKVD = kNKV * kHD;       // 1024
constexpr float kScale = 0.08838834764831845f;  // 128^-0.5
}

__device__ __forceinline__ v16h pack16(v8h lo, v8h hi) {
  union { v16h v; v8h h[2]; } u;
  u.h[0] = lo; u.h[1] = hi;
  return u.v;
}

__device__ __forceinline__ v8f wmma_f16(v16h a, v16h b, v8f c) {
  // D = A(16x32 f16) * B(32x16 f16) + C(16x16 f32)
  return __builtin_amdgcn_wmma_f32_16x16x32_f16(false, a, false, b, (short)0, c,
                                                false, false);
}

__device__ __forceinline__ float rowmax16(float v) {
  v = fmaxf(v, __shfl_xor(v, 1, 16));
  v = fmaxf(v, __shfl_xor(v, 2, 16));
  v = fmaxf(v, __shfl_xor(v, 4, 16));
  v = fmaxf(v, __shfl_xor(v, 8, 16));
  return v;
}
__device__ __forceinline__ float rowsum16(float v) {
  v += __shfl_xor(v, 1, 16);
  v += __shfl_xor(v, 2, 16);
  v += __shfl_xor(v, 4, 16);
  v += __shfl_xor(v, 8, 16);
  return v;
}

// Copy 16 halves (32B) global -> LDS, either via CDNA5 async-DMA (ASYNCcnt)
// or via a register round-trip fallback.
__device__ __forceinline__ void stage16(const _Float16* __restrict__ g,
                                        _Float16* l) {
#ifdef USE_ASYNC_LDS
  __builtin_amdgcn_global_load_async_to_lds_b128(
      (AS1 i32x4*)(void*)g, (AS3 i32x4*)(void*)l, 0, 0);
  __builtin_amdgcn_global_load_async_to_lds_b128(
      (AS1 i32x4*)(void*)(g + 8), (AS3 i32x4*)(void*)(l + 8), 0, 0);
#else
  v8h a0 = ((const v8h*)g)[0];
  v8h a1 = ((const v8h*)g)[1];
  *(v8h*)l       = a0;
  *(v8h*)(l + 8) = a1;
#endif
}

__device__ __forceinline__ void wait_stage() {
#ifdef USE_ASYNC_LDS
#if __has_builtin(__builtin_amdgcn_s_wait_asynccnt)
  __builtin_amdgcn_s_wait_asynccnt(0);
#else
  asm volatile("s_wait_asynccnt 0" ::: "memory");
#endif
#endif
}

// ---------------------------------------------------------------- convert ---
__global__ void cvt_f32_f16(const float* __restrict__ src,
                            _Float16* __restrict__ dst, long n4) {
  long i = (long)blockIdx.x * blockDim.x + threadIdx.x;
  if (i >= n4) return;
  float4 v = ((const float4*)src)[i];
  v4h o;
  o[0] = (_Float16)v.x; o[1] = (_Float16)v.y;
  o[2] = (_Float16)v.z; o[3] = (_Float16)v.w;
  ((v4h*)dst)[i] = o;
}

// ------------------------------------------------------------------- GEMM ---
// C[M,NT] (f32 row major) = A[M,K] (f16 row major) * B[NT,K]^T (f16 row major)
//                           (+ bias[NT] when BIAS)
// Block 128x256, 8 waves (2x4), wave tile 64x64 (16 WMMA per K-step),
// K-step 32, double-buffered LDS staging (async global->LDS when available).
#define GT_BM 128
#define GT_BN 256
#define GT_BK 32
#define GT_LD 40   // padded LDS row stride in halves

template <int NT, bool BIAS>
__global__ __launch_bounds__(256)
void gemm_nt_wmma(const _Float16* __restrict__ A, const _Float16* __restrict__ Bm,
                  const float* __restrict__ bias, float* __restrict__ C, int K) {
  __shared__ _Float16 As[2][GT_BM * GT_LD];
  __shared__ _Float16 Bs[2][GT_BN * GT_LD];

  const int tid  = threadIdx.x;
  const int lane = tid & 31;
  const int wave = tid >> 5;          // 0..7
  const int wm   = wave >> 2;         // 0..1  (64 rows each)
  const int wn   = wave & 3;          // 0..3  (64 cols each)
  const int lrow = lane & 15;
  const int hv   = lane >> 4;         // lane half

  const int bm0 = blockIdx.y * GT_BM;
  const int bn0 = blockIdx.x * GT_BN;

  v8f acc[4][4];
#pragma unroll
  for (int i = 0; i < 4; ++i)
#pragma unroll
    for (int j = 0; j < 4; ++j)
      acc[i][j] = (v8f){0.f, 0.f, 0.f, 0.f, 0.f, 0.f, 0.f, 0.f};

  // staging map: thread t moves 16 halves of A and 32 of B per K-step
  const int srow = tid >> 1;            // 0..127
  const int scol = (tid & 1) * 16;      // 0 or 16
  const _Float16* Ag  = A  + (size_t)(bm0 + srow) * K + scol;
  const _Float16* Bg0 = Bm + (size_t)(bn0 + srow) * K + scol;
  const _Float16* Bg1 = Bm + (size_t)(bn0 + srow + 128) * K + scol;

  auto stage = [&](int kt, int buf) {
    const int k0 = kt * GT_BK;
    stage16(Ag  + k0, &As[buf][srow * GT_LD + scol]);
    stage16(Bg0 + k0, &Bs[buf][srow * GT_LD + scol]);
    stage16(Bg1 + k0, &Bs[buf][(srow + 128) * GT_LD + scol]);
  };

  const int nk = K / GT_BK;
  stage(0, 0);
  wait_stage();
  __syncthreads();

  for (int kt = 0; kt < nk; ++kt) {
    const int cur = kt & 1;
    if (kt + 1 < nk) stage(kt + 1, cur ^ 1);

    v16h af[4], bf[4];
#pragma unroll
    for (int i = 0; i < 4; ++i) {
      const int am = wm * 64 + i * 16 + lrow;
      v8h lo = *(const v8h*)&As[cur][am * GT_LD + hv * 8];       // K 0..7 / 8..15
      v8h hi = *(const v8h*)&As[cur][am * GT_LD + 16 + hv * 8];  // K 16..23 / 24..31
      af[i] = pack16(lo, hi);
    }
#pragma unroll
    for (int j = 0; j < 4; ++j) {
      const int bn = wn * 64 + j * 16 + lrow;
      v8h lo = *(const v8h*)&Bs[cur][bn * GT_LD + hv * 8];
      v8h hi = *(const v8h*)&Bs[cur][bn * GT_LD + 16 + hv * 8];
      bf[j] = pack16(lo, hi);
    }
#pragma unroll
    for (int i = 0; i < 4; ++i)
#pragma unroll
      for (int j = 0; j < 4; ++j)
        acc[i][j] = wmma_f16(af[i], bf[j], acc[i][j]);

    wait_stage();
    __syncthreads();
  }

  // epilogue: row = tile_row + r + 8*hv, col = tile_col + lrow
#pragma unroll
  for (int j = 0; j < 4; ++j) {
    const int col = bn0 + wn * 64 + j * 16 + lrow;
    const float bv = BIAS ? bias[col] : 0.f;
#pragma unroll
    for (int i = 0; i < 4; ++i) {
      const int rowb = bm0 + wm * 64 + i * 16 + hv * 8;
      float* cp = C + (size_t)rowb * NT + col;
#pragma unroll
      for (int r = 0; r < 8; ++r)
        cp[(size_t)r * NT] = acc[i][j][r] + bv;   // NT const -> imm offsets
    }
  }
}

// ------------------------------------------------------------------- RoPE ---
// src: [B, S, nheads*HD] f32  ->  dst: [B, nheads, S, HD] f16 (rotated)
__global__ void rope_kernel(const float* __restrict__ src,
                            const float* __restrict__ cosb,
                            const float* __restrict__ sinb,
                            _Float16* __restrict__ dst, int nheads) {
  const long total = (long)kB * nheads * kS * (kHD / 2);
  long idx = (long)blockIdx.x * blockDim.x + threadIdx.x;
  if (idx >= total) return;
  const int d = (int)(idx & 63);
  long t = idx >> 6;
  const int s = (int)(t % kS); t /= kS;
  const int h = (int)(t % nheads);
  const int b = (int)(t / nheads);

  const float* srow = src + ((size_t)(b * kS + s)) * nheads * kHD + (size_t)h * kHD;
  const float x1 = srow[d];
  const float x2 = srow[d + 64];
  const float* crow = cosb + ((size_t)(b * kS + s)) * kHD;
  const float* sr   = sinb + ((size_t)(b * kS + s)) * kHD;
  const float o1 = x1 * crow[d]      - x2 * sr[d];
  const float o2 = x2 * crow[d + 64] + x1 * sr[d + 64];
  _Float16* drow = dst + (((size_t)b * nheads + h) * kS + s) * kHD;
  drow[d]      = (_Float16)o1;
  drow[d + 64] = (_Float16)o2;
}

// src: [B, S, NKV*HD] f32  ->  dst: [B, NKV, HD, S] f16 (transposed for PV)
__global__ void v_transpose(const float* __restrict__ src,
                            _Float16* __restrict__ dst) {
  const long total = (long)kB * kNKV * kS * kHD;
  long idx = (long)blockIdx.x * blockDim.x + threadIdx.x;
  if (idx >= total) return;
  const int d = (int)(idx % kHD);
  long t = idx / kHD;
  const int s = (int)(t % kS); t /= kS;
  const int h = (int)(t % kNKV);
  const int b = (int)(t / kNKV);
  const float v = src[((size_t)(b * kS + s)) * kNKVD + (size_t)h * kHD + d];
  dst[(((size_t)b * kNKV + h) * kHD + d) * kS + s] = (_Float16)v;
}

// -------------------------------------------------------- flash attention ---
// Q:[B,NH,S,HD] f16, K:[B,NKV,S,HD] f16, Vt:[B,NKV,HD,S] f16
// O:[B,S,NH*HD] f16. 4 waves/block, 16 query rows per wave, 32-key tiles.
__global__ __launch_bounds__(128)
void flash_attn(const _Float16* __restrict__ Q, const _Float16* __restrict__ Kc,
                const _Float16* __restrict__ Vt, _Float16* __restrict__ O) {
  __shared__ _Float16 Pbuf[4][16 * GT_LD];   // per-wave P staging (C->A layout)

  const int tid  = threadIdx.x;
  const int wave = tid >> 5;
  const int lane = tid & 31;
  const int lrow = lane & 15;
  const int hv   = lane >> 4;

  int bid = blockIdx.x;
  const int qt = bid % (kS / 64); bid /= (kS / 64);
  const int h  = bid % kNH;
  const int b  = bid / kNH;
  const int kvh = h / kGroups;
  const int qbase = qt * 64 + wave * 16;

  const _Float16* Qp = Q  + (((size_t)b * kNH  + h)   * kS) * kHD;
  const _Float16* Kp = Kc + (((size_t)b * kNKV + kvh) * kS) * kHD;
  const _Float16* Vp = Vt + (((size_t)b * kNKV + kvh) * kHD) * kS;

  // Q fragments: 4 x (16x32) along HD
  v16h qf[4];
#pragma unroll
  for (int f = 0; f < 4; ++f) {
    const _Float16* p = Qp + (size_t)(qbase + lrow) * kHD + f * 32;
    v8h lo = *(const v8h*)(p + hv * 8);
    v8h hi = *(const v8h*)(p + 16 + hv * 8);
    qf[f] = pack16(lo, hi);
  }

  v8f oacc[8];
#pragma unroll
  for (int t = 0; t < 8; ++t)
    oacc[t] = (v8f){0.f, 0.f, 0.f, 0.f, 0.f, 0.f, 0.f, 0.f};
  float rmax[8], rsum[8];
#pragma unroll
  for (int r = 0; r < 8; ++r) { rmax[r] = -3.0e38f; rsum[r] = 0.f; }

  _Float16* Prow = Pbuf[wave];
  const int nkt = (qbase + 16 + 31) >> 5;   // causal: keys 0 .. qbase+15

  for (int kt = 0; kt < nkt; ++kt) {
    const int kb = kt * 32;

    // S = Q K^T : two 16x16 score tiles, each chaining 4 WMMAs over HD
    v8f sc[2];
#pragma unroll
    for (int j = 0; j < 2; ++j) {
      sc[j] = (v8f){0.f, 0.f, 0.f, 0.f, 0.f, 0.f, 0.f, 0.f};
#pragma unroll
      for (int f = 0; f < 4; ++f) {
        const _Float16* p =
            Kp + (size_t)(kb + j * 16 + lrow) * kHD + f * 32;
        v8h lo = *(const v8h*)(p + hv * 8);
        v8h hi = *(const v8h*)(p + 16 + hv * 8);
        sc[j] = wmma_f16(qf[f], pack16(lo, hi), sc[j]);
      }
    }

    // scale + causal mask (row = qbase + r + 8*hv, key = kb + j*16 + lrow)
#pragma unroll
    for (int j = 0; j < 2; ++j) {
      const int key = kb + j * 16 + lrow;
      const int rowb = qbase + hv * 8;
#pragma unroll
      for (int r = 0; r < 8; ++r) {
        const float v = sc[j][r] * kScale;
        sc[j][r] = (key <= rowb + r) ? v : -1.0e30f;
      }
    }

    // online softmax stats
    float corr[8];
#pragma unroll
    for (int r = 0; r < 8; ++r) {
      float m = fmaxf(sc[0][r], sc[1][r]);
      m = rowmax16(m);
      const float nm = fmaxf(rmax[r], m);
      corr[r] = __expf(rmax[r] - nm);
      rmax[r] = nm;
    }

    // P = exp(S - m); stage to LDS in A-fragment-friendly layout
#pragma unroll
    for (int j = 0; j < 2; ++j)
#pragma unroll
      for (int r = 0; r < 8; ++r) {
        const float p = __expf(sc[j][r] - rmax[r]);
        sc[j][r] = p;
        Prow[(r + hv * 8) * GT_LD + j * 16 + lrow] = (_Float16)p;
      }

#pragma unroll
    for (int r = 0; r < 8; ++r) {
      float s = rowsum16(sc[0][r] + sc[1][r]);
      rsum[r] = rsum[r] * corr[r] + s;
    }

    // rescale running output
#pragma unroll
    for (int t = 0; t < 8; ++t)
#pragma unroll
      for (int r = 0; r < 8; ++r)
        oacc[t][r] *= corr[r];

    // same-wave LDS RAW: ensure P stores land before fragment reload
    asm volatile("s_wait_dscnt 0" ::: "memory");

    v16h pf;
    {
      const _Float16* p = &Prow[lrow * GT_LD];
      v8h lo = *(const v8h*)(p + hv * 8);
      v8h hi = *(const v8h*)(p + 16 + hv * 8);
      pf = pack16(lo, hi);
    }

    // O += P V : 8 WMMAs across the 128 head dims (Vt rows contiguous in S)
#pragma unroll
    for (int t = 0; t < 8; ++t) {
      const _Float16* p = Vp + (size_t)(t * 16 + lrow) * kS + kb;
      v8h lo = *(const v8h*)(p + hv * 8);
      v8h hi = *(const v8h*)(p + 16 + hv * 8);
      oacc[t] = wmma_f16(pf, pack16(lo, hi), oacc[t]);
    }
  }

  // normalize + store: O[b, s, h*HD + t*16 + lrow]
  float rinv[8];
#pragma unroll
  for (int r = 0; r < 8; ++r) rinv[r] = 1.f / rsum[r];

  _Float16* Op = O + ((size_t)b * kS) * kNQ + (size_t)h * kHD;
#pragma unroll
  for (int t = 0; t < 8; ++t)
#pragma unroll
    for (int r = 0; r < 8; ++r) {
      const int s = qbase + r + 8 * hv;
      const float v = oacc[t][r] * rinv[r];
      Op[(size_t)s * kNQ + t * 16 + lrow] = (_Float16)v;
    }
}

// ---------------------------------------------------------------- launch ----
extern "C" void kernel_launch(void* const* d_in, const int* in_sizes, int n_in,
                              void* d_out, int out_size, void* d_ws, size_t ws_size,
                              hipStream_t stream) {
  (void)in_sizes; (void)n_in; (void)out_size; (void)ws_size;
  const float* hs   = (const float*)d_in[0];
  const float* cosb = (const float*)d_in[1];
  const float* sinb = (const float*)d_in[2];
  const float* qw   = (const float*)d_in[3];
  const float* qb   = (const float*)d_in[4];
  const float* kw   = (const float*)d_in[5];
  const float* kbv  = (const float*)d_in[6];
  const float* vw   = (const float*)d_in[7];
  const float* vb   = (const float*)d_in[8];
  const float* ow   = (const float*)d_in[9];
  float* out = (float*)d_out;
  char* ws = (char*)d_ws;

  // workspace layout (all 256B aligned; ~298 MB total with slot reuse)
  constexpr size_t SZ_XH = (size_t)kM * kD * 2;        // X f16
  constexpr size_t SZ_WQ = (size_t)kNQ * kD * 2;       // q_w f16 (later o_w f16)
  constexpr size_t SZ_WK = (size_t)kNKVD * kD * 2;     // k_w / v_w f16
  constexpr size_t SZ_QF = (size_t)kM * kNQ * 4;       // Q f32 (later attn-out f16)
  constexpr size_t SZ_KF = (size_t)kM * kNKVD * 4;     // K / V f32
  constexpr size_t SZ_QH = (size_t)kM * kNQ * 2;       // Q f16 rotated
  constexpr size_t SZ_KH = (size_t)kM * kNKVD * 2;     // K f16 rotated / V^T f16

  constexpr size_t OFF_XH = 0;
  constexpr size_t OFF_WQ = OFF_XH + SZ_XH;
  constexpr size_t OFF_WK = OFF_WQ + SZ_WQ;
  constexpr size_t OFF_WV = OFF_WK + SZ_WK;
  constexpr size_t OFF_QF = OFF_WV + SZ_WK;
  constexpr size_t OFF_KF = OFF_QF + SZ_QF;
  constexpr size_t OFF_VF = OFF_KF + SZ_KF;
  constexpr size_t OFF_QH = OFF_VF + SZ_KF;
  constexpr size_t OFF_KH = OFF_QH + SZ_QH;
  constexpr size_t OFF_VT = OFF_KH + SZ_KH;
  constexpr size_t OFF_AO = OFF_QF;   // attn-out f16 reuses dead Q-f32 slot
  constexpr size_t OFF_WO = OFF_WQ;   // o_w f16 reuses dead q_w-f16 slot

  _Float16* Xh  = (_Float16*)(ws + OFF_XH);
  _Float16* Wqh = (_Float16*)(ws + OFF_WQ);
  _Float16* Wkh = (_Float16*)(ws + OFF_WK);
  _Float16* Wvh = (_Float16*)(ws + OFF_WV);
  float*    Qf  = (float*)   (ws + OFF_QF);
  float*    Kf  = (float*)   (ws + OFF_KF);
  float*    Vf  = (float*)   (ws + OFF_VF);
  _Float16* Qh  = (_Float16*)(ws + OFF_QH);
  _Float16* Kh  = (_Float16*)(ws + OFF_KH);
  _Float16* Vth = (_Float16*)(ws + OFF_VT);
  _Float16* AOh = (_Float16*)(ws + OFF_AO);
  _Float16* Woh = (_Float16*)(ws + OFF_WO);

  auto cvt = [&](const float* s, _Float16* d, long n) {
    long n4 = n / 4;
    cvt_f32_f16<<<(unsigned)((n4 + 255) / 256), 256, 0, stream>>>(s, d, n4);
  };

  // 1) f16 conversions of activations + QKV weights
  cvt(hs, Xh, (long)kM * kD);
  cvt(qw, Wqh, (long)kNQ * kD);
  cvt(kw, Wkh, (long)kNKVD * kD);
  cvt(vw, Wvh, (long)kNKVD * kD);

  // 2) QKV projections (NT WMMA GEMM, double-buffered async staging)
  gemm_nt_wmma<kNQ, true><<<dim3(kNQ / GT_BN, kM / GT_BM), 256, 0, stream>>>(
      Xh, Wqh, qb, Qf, kD);
  gemm_nt_wmma<kNKVD, true><<<dim3(kNKVD / GT_BN, kM / GT_BM), 256, 0, stream>>>(
      Xh, Wkh, kbv, Kf, kD);
  gemm_nt_wmma<kNKVD, true><<<dim3(kNKVD / GT_BN, kM / GT_BM), 256, 0, stream>>>(
      Xh, Wvh, vb, Vf, kD);

  // 3) RoPE + layout transforms to f16 attention operands
  {
    long n = (long)kB * kNH * kS * (kHD / 2);
    rope_kernel<<<(unsigned)((n + 255) / 256), 256, 0, stream>>>(Qf, cosb, sinb, Qh, kNH);
    n = (long)kB * kNKV * kS * (kHD / 2);
    rope_kernel<<<(unsigned)((n + 255) / 256), 256, 0, stream>>>(Kf, cosb, sinb, Kh, kNKV);
    n = (long)kB * kNKV * kS * kHD;
    v_transpose<<<(unsigned)((n + 255) / 256), 256, 0, stream>>>(Vf, Vth);
  }

  // 4) o_w -> f16 (reuses q_w slot, which is dead after the Q GEMM)
  cvt(ow, Woh, (long)kD * kNQ);

  // 5) causal GQA flash attention (WMMA)
  flash_attn<<<kB * kNH * (kS / 64), 128, 0, stream>>>(Qh, Kh, Vth, AOh);

  // 6) output projection -> f32 d_out
  gemm_nt_wmma<kD, false><<<dim3(kD / GT_BN, kM / GT_BM), 256, 0, stream>>>(
      AOh, Woh, nullptr, out, kNQ);
}
